// ODEBlock_61229053772146
// MI455X (gfx1250) — matrix operand order repaired and checked by reading, not compiled, not run on hardware
//
#include <hip/hip_runtime.h>
#include <math.h>

// ---------------- CDNA5 WMMA types ----------------
typedef __attribute__((ext_vector_type(16))) __bf16 v16bf;
typedef __attribute__((ext_vector_type(8)))  float  v8f;

union Frag16 {            // one 16x16x32 bf16 WMMA operand fragment per lane
    uint4 q[2];           // two 16-byte loads
    v16bf v;              // 32 bytes = 16 bf16
};

// Problem constants (B=1024, D=512, H=2048, 16 RK4 steps on [0,1])
#define BATCH   1024
#define DDIM    512
#define HDIM    2048
#define NSTEPS  16
#define BROWS   16                  // rows (M) per workgroup
#define YPITCH  (DDIM + 8)          // +8 bf16 pad -> conflict-free ds_load_b128
#define HPITCH  (HDIM + 8)

// -------------------------------------------------------------------------
// Prep: convert W1 [512][2048] f32 -> W1t [2048][512] bf16 (N-major, K-contig)
//       convert W2 [2048][512] f32 -> W2t [512][2048] bf16
// -------------------------------------------------------------------------
__global__ void prep_weights(const float* __restrict__ W1,
                             const float* __restrict__ W2,
                             __bf16* __restrict__ W1t,
                             __bf16* __restrict__ W2t)
{
    size_t idx = (size_t)blockIdx.x * blockDim.x + threadIdx.x;
    const size_t n1 = (size_t)DDIM * HDIM;   // 1M elements each
    if (idx < n1) {
        int n = (int)(idx / DDIM);           // 0..2047
        int k = (int)(idx % DDIM);           // 0..511
        W1t[idx] = (__bf16)W1[(size_t)k * HDIM + n];
    } else if (idx < 2 * n1) {
        size_t j = idx - n1;
        int n = (int)(j / HDIM);             // 0..511
        int k = (int)(j % HDIM);             // 0..2047
        W2t[j] = (__bf16)W2[(size_t)k * DDIM + n];
    }
}

// -------------------------------------------------------------------------
// Persistent RK4 kernel: each workgroup owns 16 batch rows and runs the full
// 16-step RK4 integration. State ping-pongs through LDS (bf16 A-operands);
// k/acc/y live in registers in WMMA C-fragment layout.
// WMMA scheduling: 4 independent accumulators interleaved in both GEMMs so
// consecutive v_wmma ops never chain on the same D registers (covers the
// 1+4-coexec bf16 WMMA RAW hazard window with in-wave ILP).
// -------------------------------------------------------------------------
__global__ __launch_bounds__(256) void ode_rk4_kernel(
    const float*  __restrict__ x,
    const __bf16* __restrict__ W1t,   // [HDIM][DDIM] bf16
    const float*  __restrict__ b1,
    const __bf16* __restrict__ W2t,   // [DDIM][HDIM] bf16
    const float*  __restrict__ b2,
    float*        __restrict__ out)
{
    extern __shared__ __align__(16) unsigned char smem[];
    __bf16* ys = (__bf16*)smem;                                     // [16][YPITCH]
    __bf16* hb = (__bf16*)(smem + BROWS * YPITCH * sizeof(__bf16)); // [16][HPITCH]

    const int tid  = threadIdx.x;
    const int wave = tid >> 5;        // 0..7
    const int lane = tid & 31;
    const int l16  = lane & 15;
    const int lhi  = lane >> 4;       // 0 or 1
    const int r0   = blockIdx.x * BROWS;

    const int crow0 = lhi * 8;        // C-fragment: lanes 0-15 -> M=i, 16-31 -> M=8+i
    const int nb1   = wave * 256;     // this wave's GEMM1 (H) columns
    const int nb2   = wave * 64;      // this wave's GEMM2 (D) columns

    // b2 bias values for this lane's 4 GEMM2 N-tiles (register resident)
    float b2r[4];
#pragma unroll
    for (int t = 0; t < 4; ++t) b2r[t] = b2[nb2 + t * 16 + l16];

    // y and RK4 accumulator in C-fragment layout: [tile][vgpr-row]
    float yreg[4][8], accr[4][8];
#pragma unroll
    for (int t = 0; t < 4; ++t)
#pragma unroll
        for (int i = 0; i < 8; ++i) {
            const int m = crow0 + i;
            const int n = nb2 + t * 16 + l16;
            float v = x[(size_t)(r0 + m) * DDIM + n];
            yreg[t][i] = v;
            accr[t][i] = 0.0f;
            ys[m * YPITCH + n] = (__bf16)v;     // stage-0 input for step 0
        }

    const float dt = (float)(1.0 / NSTEPS);

#pragma unroll 1
    for (int step = 0; step < NSTEPS; ++step) {
        const float tbase = dt * (float)step;

#pragma unroll 1
        for (int s = 0; s < 4; ++s) {
            __syncthreads();   // ystage valid for all waves; h free to overwrite
            const float tcur = tbase + (s == 0 ? 0.0f : (s == 3 ? dt : 0.5f * dt));

            // ===== GEMM1: h = tanh(ystage @ W1 + b1 + t), this wave's 256 cols
            // Hoist all 16 A fragments (K = 512) -> reused by every N-tile.
            Frag16 afr[16];
#pragma unroll
            for (int ks = 0; ks < 16; ++ks) {
                const __bf16* p = ys + l16 * YPITCH + ks * 32 + lhi * 16;
                afr[ks].q[0] = *(const uint4*)(p);
                afr[ks].q[1] = *(const uint4*)(p + 8);
            }

            // 4 N-tiles per group: 4 independent accumulator chains.
#pragma unroll 1
            for (int tg = 0; tg < 16; tg += 4) {
                const int n0 = nb1 + (tg + 0) * 16 + l16;
                const int n1 = nb1 + (tg + 1) * 16 + l16;
                const int n2 = nb1 + (tg + 2) * 16 + l16;
                const int n3 = nb1 + (tg + 3) * 16 + l16;
                const __bf16* bp0 = W1t + (size_t)n0 * DDIM + lhi * 16;
                const __bf16* bp1 = W1t + (size_t)n1 * DDIM + lhi * 16;
                const __bf16* bp2 = W1t + (size_t)n2 * DDIM + lhi * 16;
                const __bf16* bp3 = W1t + (size_t)n3 * DDIM + lhi * 16;
                v8f c0 = {}, c1 = {}, c2v = {}, c3 = {};
#pragma unroll
                for (int ks = 0; ks < 16; ++ks) {
                    Frag16 f0, f1, f2, f3;
                    f0.q[0] = *(const uint4*)(bp0 + ks * 32);
                    f0.q[1] = *(const uint4*)(bp0 + ks * 32 + 8);
                    f1.q[0] = *(const uint4*)(bp1 + ks * 32);
                    f1.q[1] = *(const uint4*)(bp1 + ks * 32 + 8);
                    f2.q[0] = *(const uint4*)(bp2 + ks * 32);
                    f2.q[1] = *(const uint4*)(bp2 + ks * 32 + 8);
                    f3.q[0] = *(const uint4*)(bp3 + ks * 32);
                    f3.q[1] = *(const uint4*)(bp3 + ks * 32 + 8);
                    c0  = __builtin_amdgcn_wmma_f32_16x16x32_bf16(
                              false, afr[ks].v, false, f0.v, (short)0, c0,  false, false);
                    c1  = __builtin_amdgcn_wmma_f32_16x16x32_bf16(
                              false, afr[ks].v, false, f1.v, (short)0, c1,  false, false);
                    c2v = __builtin_amdgcn_wmma_f32_16x16x32_bf16(
                              false, afr[ks].v, false, f2.v, (short)0, c2v, false, false);
                    c3  = __builtin_amdgcn_wmma_f32_16x16x32_bf16(
                              false, afr[ks].v, false, f3.v, (short)0, c3,  false, false);
                }
                // fused epilogue: +b1 +t, tanh, bf16 -> h LDS
                const float bias0 = b1[n0] + tcur;
                const float bias1 = b1[n1] + tcur;
                const float bias2 = b1[n2] + tcur;
                const float bias3 = b1[n3] + tcur;
#pragma unroll
                for (int i = 0; i < 8; ++i) {
                    const int m = crow0 + i;
                    hb[m * HPITCH + n0] = (__bf16)tanhf(c0[i]  + bias0);
                    hb[m * HPITCH + n1] = (__bf16)tanhf(c1[i]  + bias1);
                    hb[m * HPITCH + n2] = (__bf16)tanhf(c2v[i] + bias2);
                    hb[m * HPITCH + n3] = (__bf16)tanhf(c3[i]  + bias3);
                }
            }
            __syncthreads();   // h fully written by all waves

            // ===== GEMM2: k = h @ W2 + b2, this wave's 64 cols (4 N-tiles)
            v8f ck[4];
#pragma unroll
            for (int t = 0; t < 4; ++t) ck[t] = (v8f){};

#pragma unroll 1
            for (int kc = 0; kc < 8; ++kc) {           // 8 chunks x 8 K-steps = K 2048
                Frag16 af[8];
#pragma unroll
                for (int ks = 0; ks < 8; ++ks) {
                    const __bf16* p = hb + l16 * HPITCH + kc * 256 + ks * 32 + lhi * 16;
                    af[ks].q[0] = *(const uint4*)(p);
                    af[ks].q[1] = *(const uint4*)(p + 8);
                }
                // ks-outer / tile-inner: consecutive wmmas hit different accs
#pragma unroll
                for (int ks = 0; ks < 8; ++ks) {
#pragma unroll
                    for (int t = 0; t < 4; ++t) {
                        const int n = nb2 + t * 16 + l16;
                        const __bf16* bp = W2t + (size_t)n * HDIM + kc * 256 + lhi * 16;
                        Frag16 bf;
                        bf.q[0] = *(const uint4*)(bp + ks * 32);
                        bf.q[1] = *(const uint4*)(bp + ks * 32 + 8);
                        ck[t] = __builtin_amdgcn_wmma_f32_16x16x32_bf16(
                                    false, af[ks].v, false, bf.v, (short)0, ck[t], false, false);
                    }
                }
            }

            // ===== RK4 stage combine (registers, C-fragment layout)
            const float kws = (s == 1 || s == 2) ? 2.0f : 1.0f;
            if (s < 3) {
                const float csd = (s == 2) ? dt : 0.5f * dt;
#pragma unroll
                for (int t = 0; t < 4; ++t) {
                    const int n = nb2 + t * 16 + l16;
#pragma unroll
                    for (int i = 0; i < 8; ++i) {
                        float k = ck[t][i] + b2r[t];
                        accr[t][i] += kws * k;
                        float yn = yreg[t][i] + csd * k;
                        ys[(crow0 + i) * YPITCH + n] = (__bf16)yn;  // next stage input
                    }
                }
            } else {
                const float g = dt * (1.0f / 6.0f);
#pragma unroll
                for (int t = 0; t < 4; ++t) {
                    const int n = nb2 + t * 16 + l16;
#pragma unroll
                    for (int i = 0; i < 8; ++i) {
                        float k = ck[t][i] + b2r[t];
                        accr[t][i] += k;
                        yreg[t][i] += g * accr[t][i];
                        accr[t][i] = 0.0f;
                        ys[(crow0 + i) * YPITCH + n] = (__bf16)yreg[t][i]; // step n+1 input
                    }
                }
            }
        } // stages
    } // steps

    // Final state y(t=1) -> out (f32)
#pragma unroll
    for (int t = 0; t < 4; ++t) {
        const int n = nb2 + t * 16 + l16;
#pragma unroll
        for (int i = 0; i < 8; ++i)
            out[(size_t)(r0 + crow0 + i) * DDIM + n] = yreg[t][i];
    }
}

// -------------------------------------------------------------------------
extern "C" void kernel_launch(void* const* d_in, const int* in_sizes, int n_in,
                              void* d_out, int out_size, void* d_ws, size_t ws_size,
                              hipStream_t stream) {
    const float* x  = (const float*)d_in[0];   // [1024,512]
    const float* W1 = (const float*)d_in[1];   // [512,2048]
    const float* b1 = (const float*)d_in[2];   // [2048]
    const float* W2 = (const float*)d_in[3];   // [2048,512]
    const float* b2 = (const float*)d_in[4];   // [512]
    float* out = (float*)d_out;                // [1024,512]

    // Workspace: bf16 transposed weights (2 MB + 2 MB)
    __bf16* W1t = (__bf16*)d_ws;
    __bf16* W2t = W1t + (size_t)HDIM * DDIM;

    const size_t total = 2ull * (size_t)HDIM * DDIM;
    prep_weights<<<(unsigned)((total + 255) / 256), 256, 0, stream>>>(W1, W2, W1t, W2t);

    const size_t lds_bytes = ((size_t)BROWS * YPITCH + (size_t)BROWS * HPITCH) * sizeof(__bf16); // 82,432 B
    ode_rk4_kernel<<<BATCH / BROWS, 256, lds_bytes, stream>>>(x, W1t, b1, W2t, b2, out);
}
